// Histogram2D_31086973288713
// MI455X (gfx1250) — matrix operand order repaired and checked by reading, not compiled, not run on hardware
//
#include <hip/hip_runtime.h>
#include <math.h>

// CDNA5 / gfx1250: wave32 WMMA f16 -> f32, 16x16x32
typedef _Float16 v16h __attribute__((ext_vector_type(16)));
typedef _Float16 v8h  __attribute__((ext_vector_type(8)));
typedef float    v8f  __attribute__((ext_vector_type(8)));

#define NBINS   128
#define CHUNK   32      // points per iteration == WMMA K
#define WGSIZE  256     // 8 waves of 32
#define NTILE   (NBINS / 16)   // 8 j-tiles per wave row-slab
#define NBLOCKS 1024

// ---------------------------------------------------------------------------
// Kernel 1: zero the output accumulator (harness poisons d_out with 0xAA).
// ---------------------------------------------------------------------------
__global__ void hist_zero_kernel(float* __restrict__ out, int n) {
    int i = blockIdx.x * blockDim.x + threadIdx.x;
    if (i < n) out[i] = 0.0f;
}

// ---------------------------------------------------------------------------
// Kernel 2: windowed Gaussian-CDF weights -> LDS (f16, transposed) -> WMMA
// C[i][j] += sum_k wx[k][i] * wy[k][j]   (K = points, chunked by 32)
// ---------------------------------------------------------------------------
__global__ __launch_bounds__(WGSIZE) void hist_wmma_kernel(
    const float* __restrict__ x,
    const float* __restrict__ ex,
    const float* __restrict__ ey,
    float* __restrict__ out,
    int n)
{
    // wT[axis][bin][point] : transposed so WMMA fragments are contiguous b128s
    __shared__ __attribute__((aligned(16))) _Float16 wT[2 * NBINS * CHUNK];

    const int tid  = threadIdx.x;
    const int wave = tid >> 5;
    const int lane = tid & 31;

    // uniform per-axis grid parameters (edges are uniform linspace)
    const float e0x = ex[0], dbx = ex[1] - ex[0];
    const float e0y = ey[0], dby = ey[1] - ey[0];

    // ---- weight-phase mapping: 4 threads per (point, axis) ----
    const int p    = tid >> 3;        // point in chunk: 0..31
    const int axis = (tid >> 2) & 1;  // 0 = x-axis, 1 = y-axis
    const int q    = tid & 3;         // quarter of the 16-bin window
    const float e0    = axis ? e0y : e0x;
    const float db    = axis ? dby : dbx;
    const float invdb = 1.0f / db;
    // bw = BANDWIDTH * db = db ;  1/(bw*sqrt(2))
    const float invc  = invdb * 0.70710678118654752f;

    // ---- WMMA fragment LDS offsets (units: halves) ----
    // A (16x32 f16): lane m<16 holds row m, K {0..7}u{16..23}; lanes 16..31 hold
    //                row m, K {8..15}u{24..31}  -> two contiguous 16B loads.
    // B (32x16 f16): lanes 0..15 col n, K {0..7}u{16..23}; lanes 16..31 col n,
    //                K {8..15}u{24..31} -> two 16B loads (two v8h).
    const int m    = lane & 15;
    const int grp  = lane >> 4;
    const unsigned offA1 = (unsigned)((16 * wave + m) * CHUNK + 8 * grp);
    const unsigned offA2 = offA1 + 16;
    const unsigned offB0 = (unsigned)(m * CHUNK + 8 * grp);

    v8f acc[NTILE];
#pragma unroll
    for (int t = 0; t < NTILE; ++t) acc[t] = (v8f)0.0f;

    const int nChunks = (n + CHUNK - 1) / CHUNK;
    uint4* zp = (uint4*)wT;   // 16 KB = 1024 uint4

    for (int c = blockIdx.x; c < nChunks; c += gridDim.x) {
        // ---- phase 0: zero the weight tile (window writes are sparse) ----
#pragma unroll
        for (int k = 0; k < 4; ++k) {
            uint4 z; z.x = 0u; z.y = 0u; z.z = 0u; z.w = 0u;
            zp[tid * 4 + k] = z;
        }
        __syncthreads();

        // ---- phase 1: Gaussian mass per bin via erf-CDF differences.
        // Only a 16-bin window around the particle is non-saturated
        // (bw = 1 bin => erf saturates within ~6 bins); rest stays 0.
        {
            long gp = (long)c * CHUNK + p;
            float u = (gp < (long)n) ? x[gp * 6 + axis] : 3.0e37f;
            float fi = (u - e0) * invdb;
            fi = fminf(fmaxf(fi, -1.0e6f), 1.0e6f);
            int w0 = (int)floorf(fi) - 8 + 4 * q;   // this thread's 4 bins
            float cdf[5];
#pragma unroll
            for (int t = 0; t < 5; ++t) {
                float edge = e0 + (float)(w0 + t) * db;
                cdf[t] = erff((edge - u) * invc);
            }
            _Float16* dst = &wT[axis * (NBINS * CHUNK)];
#pragma unroll
            for (int t = 0; t < 4; ++t) {
                int b = w0 + t;
                if (0 <= b && b < NBINS)
                    dst[b * CHUNK + p] = (_Float16)(0.5f * (cdf[t + 1] - cdf[t]));
            }
        }
        __syncthreads();

        // ---- phase 2: 8 waves x 8 j-tiles cover the 128x128 output ----
        {
            const v8h* wx = (const v8h*)&wT[0];
            const v8h* wy = (const v8h*)&wT[NBINS * CHUNK];
            v8h alo = wx[offA1 >> 3];
            v8h ahi = wx[offA2 >> 3];
            v16h a = __builtin_shufflevector(alo, ahi,
                       0,1,2,3,4,5,6,7,8,9,10,11,12,13,14,15);
#pragma unroll
            for (int t = 0; t < NTILE; ++t) {
                unsigned ob = (offB0 + (unsigned)(t * 16 * CHUNK)) >> 3;
                v8h blo = wy[ob];
                v8h bhi = wy[ob + 2];
                v16h b = __builtin_shufflevector(blo, bhi,
                           0,1,2,3,4,5,6,7,8,9,10,11,12,13,14,15);
                acc[t] = __builtin_amdgcn_wmma_f32_16x16x32_f16(
                             false, a, false, b, (short)0, acc[t], false, false);
            }
        }
        __syncthreads();   // protect LDS re-zero of next chunk
    }

    // ---- write out: hardware f32 atomics into the global 128x128 grid ----
    // C/D layout: VGPR v, lane l -> row = v + 8*(l>>4), col = l&15 (per tile).
#pragma unroll
    for (int t = 0; t < NTILE; ++t) {
        int col = 16 * t + (lane & 15);
#pragma unroll
        for (int v = 0; v < 8; ++v) {
            int row = 16 * wave + v + 8 * (lane >> 4);
            unsafeAtomicAdd(&out[row * NBINS + col], acc[t][v]);
        }
    }
}

// ---------------------------------------------------------------------------
// Kernel 3: normalize: hist /= (hist.sum() * dx * dy)   (single workgroup)
// ---------------------------------------------------------------------------
__global__ __launch_bounds__(WGSIZE) void hist_finalize_kernel(
    const float* __restrict__ ex,
    const float* __restrict__ ey,
    float* __restrict__ out)
{
    __shared__ float red[WGSIZE];
    const int tid = threadIdx.x;
    const int total = NBINS * NBINS;

    float s = 0.0f;
    for (int i = tid; i < total; i += WGSIZE) s += out[i];
    red[tid] = s;
    __syncthreads();
    for (int off = WGSIZE / 2; off > 0; off >>= 1) {
        if (tid < off) red[tid] += red[tid + off];
        __syncthreads();
    }
    float dx = ex[1] - ex[0];
    float dy = ey[1] - ey[0];
    float scale = 1.0f / (red[0] * dx * dy);
    for (int i = tid; i < total; i += WGSIZE) out[i] *= scale;
}

// ---------------------------------------------------------------------------
extern "C" void kernel_launch(void* const* d_in, const int* in_sizes, int n_in,
                              void* d_out, int out_size, void* d_ws, size_t ws_size,
                              hipStream_t stream) {
    (void)n_in; (void)d_ws; (void)ws_size;
    const float* x  = (const float*)d_in[0];   // (N, 6) float32
    const float* ex = (const float*)d_in[1];   // (129,) float32
    const float* ey = (const float*)d_in[2];   // (129,) float32
    float* out = (float*)d_out;                // (128, 128) float32

    int n = in_sizes[0] / 6;                   // number of particles

    hist_zero_kernel<<<(out_size + WGSIZE - 1) / WGSIZE, WGSIZE, 0, stream>>>(out, out_size);
    hist_wmma_kernel<<<NBLOCKS, WGSIZE, 0, stream>>>(x, ex, ey, out, n);
    hist_finalize_kernel<<<1, WGSIZE, 0, stream>>>(ex, ey, out);
}
